// Interaction_Flat_34059090657898
// MI455X (gfx1250) — matrix operand order repaired and checked by reading, not compiled, not run on hardware
//
#include <hip/hip_runtime.h>
#include <math.h>

#define BATCH  32
#define MAXD   50
#define MAXP   545
#define EMB    384
#define HEADS  12
#define HD     32
#define INTER  1536
#define FLATN  78192
#define NLAYER 2

typedef __attribute__((ext_vector_type(8)))  _Float16 v8h;
typedef __attribute__((ext_vector_type(16))) _Float16 v16h;
typedef __attribute__((ext_vector_type(8)))  float    v8f;

__device__ __forceinline__ v8f wmma_f16(v16h a, v16h b, v8f c) {
  return __builtin_amdgcn_wmma_f32_16x16x32_f16(false, a, false, b, (short)0, c, false, false);
}

// ---------------------------------------------------------------------------
// WMMA GEMM: C[bz] = A[bz] (MxK row-major f16) @ Bt[bz]^T (Bt: NxK row-major
// f16) + bias, optional ReLU, f32 or f16 output. One wave computes a 16x64
// C tile: per k-step, load A (2xb128) + 4 B fragments (8xb128) as one burst
// (sched_barrier keeps them clustered -> one s_wait), then 4 back-to-back
// WMMAs on independent accumulators. Optional split-K (ksplit>1): partials
// atomically accumulated into pre-zeroed f32 C.
// ---------------------------------------------------------------------------
__global__ __launch_bounds__(32) void gemm_wmma_kernel(
    const _Float16* __restrict__ A, const _Float16* __restrict__ Bt,
    const float* __restrict__ bias, float* __restrict__ C32,
    _Float16* __restrict__ C16, int M, int N, int K,
    long sA, long sB, long sC, int relu, int ksplit)
{
  const int lane = threadIdx.x;
  const int tm = blockIdx.x, tg = blockIdx.y;
  const int bz = blockIdx.z / ksplit;
  const int ks = blockIdx.z % ksplit;
  const bool hi = lane >= 16;
  const int mm = lane & 15;

  // this slice's K range (chunk is a multiple of 32 so vector alignment holds)
  const int chunk = (((K + ksplit - 1) / ksplit) + 31) & ~31;
  const int kbeg = ks * chunk;
  if (kbeg >= K) return;                  // uniform per wave
  const int kend = min(kbeg + chunk, K);
  const int kfull = kbeg + ((kend - kbeg) & ~31);

  const _Float16* Ab = A + (long)bz * sA;
  const _Float16* Bb = Bt + (long)bz * sB;
  const int rowc = min(tm * 16 + mm, M - 1);
  const _Float16* arow = Ab + (long)rowc * K;

  const _Float16* brow0 = Bb + (long)min((tg * 4 + 0) * 16 + mm, N - 1) * K;
  const _Float16* brow1 = Bb + (long)min((tg * 4 + 1) * 16 + mm, N - 1) * K;
  const _Float16* brow2 = Bb + (long)min((tg * 4 + 2) * 16 + mm, N - 1) * K;
  const _Float16* brow3 = Bb + (long)min((tg * 4 + 3) * 16 + mm, N - 1) * K;

  v8f acc0 = {}, acc1 = {}, acc2 = {}, acc3 = {};

  for (int kk = kbeg; kk < kfull; kk += 32) {
    // ---- one load burst: 2 + 8 x global_load_b128 ----
    const _Float16* ap = arow + kk + (hi ? 16 : 0);
    v8h alo = *(const v8h*)ap;
    v8h ahi = *(const v8h*)(ap + 8);
    const int ko = kk + (hi ? 16 : 0);
    v16h b0 = *(const v16h*)(brow0 + ko);
    v16h b1 = *(const v16h*)(brow1 + ko);
    v16h b2 = *(const v16h*)(brow2 + ko);
    v16h b3 = *(const v16h*)(brow3 + ko);
    // speculative prefetch (safe if OOB on CDNA5), no guard -> no branch
    __builtin_prefetch((const void*)(arow + kk + 64), 0, 3);
    v16h a = __builtin_shufflevector(alo, ahi, 0,1,2,3,4,5,6,7,8,9,10,11,12,13,14,15);
    // keep all loads above this point: forces distinct regs for b0..b3 so the
    // loads issue as one clause with a single wait before the WMMA block
    __builtin_amdgcn_sched_barrier(0);
    // ---- dense WMMA block on independent accumulators ----
    acc0 = wmma_f16(a, b0, acc0);
    acc1 = wmma_f16(a, b1, acc1);
    acc2 = wmma_f16(a, b2, acc2);
    acc3 = wmma_f16(a, b3, acc3);
  }
  if (kfull < kend) {  // K-tail (e.g. K = 78192 -> tail of 16), guarded scalar
    v16h a;
#pragma unroll
    for (int i = 0; i < 8; i++) {
      int ka  = kfull + (hi ? 16 : 0) + i;
      int ka2 = kfull + (hi ? 24 : 8) + i;
      a[i]     = (ka  < kend) ? arow[ka]  : (_Float16)0.f;
      a[i + 8] = (ka2 < kend) ? arow[ka2] : (_Float16)0.f;
    }
    const _Float16* browt[4] = { brow0, brow1, brow2, brow3 };
    v8f* accp[4] = { &acc0, &acc1, &acc2, &acc3 };
#pragma unroll
    for (int t = 0; t < 4; t++) {
      v16h b;
#pragma unroll
      for (int j = 0; j < 16; j++) {
        int kb = kfull + (hi ? 16 : 0) + j;
        b[j] = (kb < kend) ? browt[t][kb] : (_Float16)0.f;
      }
      *accp[t] = wmma_f16(a, b, *accp[t]);
    }
  }

  const v8f accs[4] = { acc0, acc1, acc2, acc3 };
#pragma unroll
  for (int t = 0; t < 4; t++) {
    int col = (tg * 4 + t) * 16 + mm;
#pragma unroll
    for (int r = 0; r < 8; r++) {
      int row = tm * 16 + r + (hi ? 8 : 0);
      if (row < M && col < N) {
        long idx = (long)bz * sC + (long)row * N + col;
        if (ksplit == 1) {
          float v = accs[t][r] + (bias ? bias[col] : 0.f);
          if (relu) v = fmaxf(v, 0.f);
          if (C16) C16[idx] = (_Float16)v;
          else     C32[idx] = v;
        } else {
          float v = accs[t][r] + ((bias && ks == 0) ? bias[col] : 0.f);
          atomicAdd(&C32[idx], v);
        }
      }
    }
  }
}

// ---------------------------------------------------------------------------
// Flash-style attention: one wave per (qblock, head, batch). Q@K^T and P@V
// via WMMA f16; scores + probs tiles live in LDS.
// ---------------------------------------------------------------------------
__global__ __launch_bounds__(32) void attn_kernel(
    const _Float16* __restrict__ Q, const _Float16* __restrict__ Kk,
    const _Float16* __restrict__ V, const float* __restrict__ mask,
    _Float16* __restrict__ O, int L)
{
  __shared__ float    sc[16][592];
  __shared__ _Float16 pr[16][592];
  const int lane = threadIdx.x;
  const int qb = blockIdx.x, h = blockIdx.y, bb = blockIdx.z;
  const bool hi = lane >= 16;
  const int mm = lane & 15;
  const long base = (long)bb * L;

  // Q fragment (A-matrix, 16x32)
  const int qrow = min(qb * 16 + mm, L - 1);
  const _Float16* qp = Q + (base + qrow) * EMB + h * HD + (hi ? 16 : 0);
  v8h qlo = *(const v8h*)qp;
  v8h qhi = *(const v8h*)(qp + 8);
  v16h a = __builtin_shufflevector(qlo, qhi, 0,1,2,3,4,5,6,7,8,9,10,11,12,13,14,15);

  const int nKB = (L + 15) / 16;
  const float scale = 0.17677669529663687f;  // 1/sqrt(HD)
  for (int kb = 0; kb < nKB; kb++) {
    int key = kb * 16 + mm;
    int keyc = min(key, L - 1);
    v16h bf = *(const v16h*)(Kk + (base + keyc) * EMB + h * HD + (hi ? 16 : 0));
    v8f c = {};
    c = wmma_f16(a, bf, c);
    float exm = (key < L) ? (1.0f - mask[bb * L + key]) * -10000.0f : 0.0f;
#pragma unroll
    for (int r = 0; r < 8; r++) {
      int rr = r + (hi ? 8 : 0);
      sc[rr][key] = (key < L) ? (c[r] * scale + exm) : -1e30f;
    }
  }
  __syncthreads();

  const int PADK = ((L + 31) / 32) * 32;
  if (lane < 16) {
    float mx = -1e30f;
    for (int k2 = 0; k2 < L; k2++) mx = fmaxf(mx, sc[lane][k2]);
    float s = 0.f;
    for (int k2 = 0; k2 < L; k2++) s += __expf(sc[lane][k2] - mx);
    float inv = 1.f / s;
    for (int k2 = 0; k2 < PADK; k2++)
      pr[lane][k2] = (k2 < L) ? (_Float16)(__expf(sc[lane][k2] - mx) * inv)
                              : (_Float16)0.f;
  }
  __syncthreads();

  v8f c0 = {}, c1 = {};
  for (int kb = 0; kb * 32 < PADK; kb++) {
    v16h ap;
#pragma unroll
    for (int i = 0; i < 8; i++) {
      ap[i]     = pr[mm][kb * 32 + (hi ? 16 : 0) + i];
      ap[i + 8] = pr[mm][kb * 32 + (hi ? 24 : 8) + i];
    }
    const int k0 = kb * 32 + (hi ? 16 : 0);
    v16h b0, b1;
#pragma unroll
    for (int j = 0; j < 16; j++) {
      int key = min(k0 + j, L - 1);  // probs are zero-padded, clamp is safe
      const _Float16* vp = V + (base + key) * EMB + h * HD;
      b0[j] = vp[mm];
      b1[j] = vp[16 + mm];
    }
    c0 = wmma_f16(ap, b0, c0);
    c1 = wmma_f16(ap, b1, c1);
  }
#pragma unroll
  for (int r = 0; r < 8; r++) {
    int qq = qb * 16 + r + (hi ? 8 : 0);
    if (qq < L) {
      long ob = (base + qq) * EMB + h * HD;
      O[ob + mm]      = (_Float16)c0[r];
      O[ob + 16 + mm] = (_Float16)c1[r];
    }
  }
}

// ---------------------------------------------------------------------------
// Embedding gather + LayerNorm -> fp32 + f16 copies. One block per row.
// ---------------------------------------------------------------------------
__global__ void embed_ln_kernel(const int* __restrict__ ids,
                                const float* __restrict__ word,
                                const float* __restrict__ pos,
                                const float* __restrict__ g,
                                const float* __restrict__ bta,
                                float* __restrict__ x32,
                                _Float16* __restrict__ x16, int L)
{
  const int row = blockIdx.x;
  const int l = row % L;
  const int id = ids[row];
  __shared__ float buf[EMB];
  __shared__ float red[128];
  const int t = threadIdx.x;
  float local = 0.f;
  for (int j = t; j < EMB; j += 128) {
    float e = word[(long)id * EMB + j] + pos[l * EMB + j];
    buf[j] = e; local += e;
  }
  red[t] = local; __syncthreads();
  for (int s1 = 64; s1 > 0; s1 >>= 1) { if (t < s1) red[t] += red[t + s1]; __syncthreads(); }
  float mean = red[0] / EMB; __syncthreads();
  local = 0.f;
  for (int j = t; j < EMB; j += 128) { float d = buf[j] - mean; local += d * d; }
  red[t] = local; __syncthreads();
  for (int s1 = 64; s1 > 0; s1 >>= 1) { if (t < s1) red[t] += red[t + s1]; __syncthreads(); }
  float inv = rsqrtf(red[0] / EMB + 1e-12f); __syncthreads();
  for (int j = t; j < EMB; j += 128) {
    float v = (buf[j] - mean) * inv * g[j] + bta[j];
    x32[(long)row * EMB + j] = v;
    x16[(long)row * EMB + j] = (_Float16)v;
  }
}

// x = LN(x + y) -> fp32 (in place) + f16 copy
__global__ void residual_ln_kernel(float* __restrict__ x32,
                                   const float* __restrict__ y32,
                                   const float* __restrict__ g,
                                   const float* __restrict__ bta,
                                   _Float16* __restrict__ x16)
{
  const int row = blockIdx.x;
  __shared__ float buf[EMB];
  __shared__ float red[128];
  const int t = threadIdx.x;
  float local = 0.f;
  for (int j = t; j < EMB; j += 128) {
    float e = x32[(long)row * EMB + j] + y32[(long)row * EMB + j];
    buf[j] = e; local += e;
  }
  red[t] = local; __syncthreads();
  for (int s1 = 64; s1 > 0; s1 >>= 1) { if (t < s1) red[t] += red[t + s1]; __syncthreads(); }
  float mean = red[0] / EMB; __syncthreads();
  local = 0.f;
  for (int j = t; j < EMB; j += 128) { float d = buf[j] - mean; local += d * d; }
  red[t] = local; __syncthreads();
  for (int s1 = 64; s1 > 0; s1 >>= 1) { if (t < s1) red[t] += red[t + s1]; __syncthreads(); }
  float inv = rsqrtf(red[0] / EMB + 1e-12f); __syncthreads();
  for (int j = t; j < EMB; j += 128) {
    float v = (buf[j] - mean) * inv * g[j] + bta[j];
    x32[(long)row * EMB + j] = v;
    x16[(long)row * EMB + j] = (_Float16)v;
  }
}

// Training-mode BatchNorm over batch axis (32 rows), optional ReLU on input,
// f16 output
__global__ void bn_kernel(const float* __restrict__ in, const float* __restrict__ g,
                          const float* __restrict__ bta, _Float16* __restrict__ out,
                          int N, int relu_in)
{
  const int n = blockIdx.x * blockDim.x + threadIdx.x;
  if (n >= N) return;
  float m = 0.f;
  for (int r = 0; r < BATCH; r++) {
    float x = in[r * N + n];
    if (relu_in) x = fmaxf(x, 0.f);
    m += x;
  }
  m *= (1.f / BATCH);
  float v = 0.f;
  for (int r = 0; r < BATCH; r++) {
    float x = in[r * N + n];
    if (relu_in) x = fmaxf(x, 0.f);
    float d = x - m; v += d * d;
  }
  v *= (1.f / BATCH);
  float inv = rsqrtf(v + 1e-5f);
  for (int r = 0; r < BATCH; r++) {
    float x = in[r * N + n];
    if (relu_in) x = fmaxf(x, 0.f);
    out[r * N + n] = (_Float16)((x - m) * inv * g[n] + bta[n]);
  }
}

// conv2d(1->3, k=3, valid) over (B,50,545) -> flattened (B,78192) f16
__global__ void conv3_kernel(const float* __restrict__ I, const float* __restrict__ w,
                             const float* __restrict__ cb, _Float16* __restrict__ out)
{
  const int total = BATCH * 3 * 48 * 543;
  int t = blockIdx.x * blockDim.x + threadIdx.x;
  if (t >= total) return;
  int j = t % 543; int tmp = t / 543;
  int i = tmp % 48; tmp /= 48;
  int o = tmp % 3;  int bb = tmp / 3;
  const float* Ib = I + (long)bb * MAXD * MAXP;
  float s = cb[o];
#pragma unroll
  for (int ki = 0; ki < 3; ki++)
#pragma unroll
    for (int kj = 0; kj < 3; kj++)
      s += Ib[(i + ki) * MAXP + (j + kj)] * w[o * 9 + ki * 3 + kj];
  out[(long)bb * FLATN + (o * 48 + i) * 543 + j] = (_Float16)s;
}

// transpose-convert: src (K,N) f32 row-major -> dst (N,K) f16 row-major
__global__ void transp_f16_kernel(const float* __restrict__ src,
                                  _Float16* __restrict__ dst, int K, int N)
{
  long t = (long)blockIdx.x * blockDim.x + threadIdx.x;
  long total = (long)K * N;
  if (t >= total) return;
  long k = t / N, n = t % N;
  dst[n * K + k] = (_Float16)src[t];
}

// ---------------------------------------------------------------------------
// Host-side helpers
// ---------------------------------------------------------------------------
static void launch_gemm(hipStream_t s, const _Float16* A, const _Float16* Bt,
                        const float* bias, float* C32, _Float16* C16,
                        int M, int N, int K, int batch,
                        long sA, long sB, long sC, int relu, int ksplit = 1)
{
  dim3 g((M + 15) / 16, (N + 63) / 64, batch * ksplit);
  gemm_wmma_kernel<<<g, 32, 0, s>>>(A, Bt, bias, C32, C16, M, N, K,
                                    sA, sB, sC, relu, ksplit);
}

static void launch_transp(hipStream_t s, const float* src, _Float16* dst, int K, int N)
{
  long total = (long)K * N;
  int blocks = (int)((total + 255) / 256);
  transp_f16_kernel<<<blocks, 256, 0, s>>>(src, dst, K, N);
}

static void run_encoder(hipStream_t s, const float* const* ep,
                        float* x32, _Float16* x16, int L, const float* mask,
                        _Float16* wqT, _Float16* wkT, _Float16* wvT, _Float16* woT,
                        _Float16* wiT, _Float16* wo2T,
                        _Float16* q16, _Float16* k16, _Float16* v16, _Float16* ctx16,
                        float* y32, _Float16* h16)
{
  const int Mrows = BATCH * L;
  for (int l = 0; l < NLAYER; l++) {
    const float* Wq  = ep[0]  + (size_t)l * EMB * EMB;   const float* bq  = ep[1]  + l * EMB;
    const float* Wk  = ep[2]  + (size_t)l * EMB * EMB;   const float* bk  = ep[3]  + l * EMB;
    const float* Wv  = ep[4]  + (size_t)l * EMB * EMB;   const float* bv  = ep[5]  + l * EMB;
    const float* Wo  = ep[6]  + (size_t)l * EMB * EMB;   const float* bo  = ep[7]  + l * EMB;
    const float* g1  = ep[8]  + l * EMB;                 const float* b1  = ep[9]  + l * EMB;
    const float* Wi  = ep[10] + (size_t)l * EMB * INTER; const float* bi  = ep[11] + l * INTER;
    const float* Wo2 = ep[12] + (size_t)l * INTER * EMB; const float* bo2 = ep[13] + l * EMB;
    const float* g2  = ep[14] + l * EMB;                 const float* b2  = ep[15] + l * EMB;

    launch_transp(s, Wq, wqT, EMB, EMB);
    launch_transp(s, Wk, wkT, EMB, EMB);
    launch_transp(s, Wv, wvT, EMB, EMB);
    launch_transp(s, Wo, woT, EMB, EMB);
    launch_transp(s, Wi, wiT, EMB, INTER);
    launch_transp(s, Wo2, wo2T, INTER, EMB);

    launch_gemm(s, x16, wqT, bq, nullptr, q16, Mrows, EMB, EMB, 1, 0, 0, 0, 0);
    launch_gemm(s, x16, wkT, bk, nullptr, k16, Mrows, EMB, EMB, 1, 0, 0, 0, 0);
    launch_gemm(s, x16, wvT, bv, nullptr, v16, Mrows, EMB, EMB, 1, 0, 0, 0, 0);

    dim3 ga((L + 15) / 16, HEADS, BATCH);
    attn_kernel<<<ga, 32, 0, s>>>(q16, k16, v16, mask, ctx16, L);

    launch_gemm(s, ctx16, woT, bo, y32, nullptr, Mrows, EMB, EMB, 1, 0, 0, 0, 0);
    residual_ln_kernel<<<Mrows, 128, 0, s>>>(x32, y32, g1, b1, x16);

    launch_gemm(s, x16, wiT, bi, nullptr, h16, Mrows, INTER, EMB, 1, 0, 0, 0, 1);
    launch_gemm(s, h16, wo2T, bo2, y32, nullptr, Mrows, EMB, INTER, 1, 0, 0, 0, 0);
    residual_ln_kernel<<<Mrows, 128, 0, s>>>(x32, y32, g2, b2, x16);
  }
}

// ---------------------------------------------------------------------------
extern "C" void kernel_launch(void* const* d_in, const int* in_sizes, int n_in,
                              void* d_out, int out_size, void* d_ws, size_t ws_size,
                              hipStream_t stream)
{
  (void)in_sizes; (void)n_in; (void)out_size; (void)ws_size;
#define FIN(i) ((const float*)d_in[(i)])
  const int* d_ids = (const int*)d_in[0];
  const int* p_ids = (const int*)d_in[1];
  const float* d_mask = FIN(2);
  const float* p_mask = FIN(3);
  const float* d_word = FIN(4);  const float* d_pos = FIN(5);
  const float* d_lng  = FIN(6);  const float* d_lnb = FIN(7);
  const float* p_word = FIN(8);  const float* p_pos = FIN(9);
  const float* p_lng  = FIN(10); const float* p_lnb = FIN(11);
  const float* denc[16]; for (int j = 0; j < 16; j++) denc[j] = FIN(12 + j);
  const float* penc[16]; for (int j = 0; j < 16; j++) penc[j] = FIN(28 + j);
  const float* conv_w = FIN(44); const float* conv_b = FIN(45);
  const float* W1 = FIN(46); const float* b1 = FIN(47);
  const float* bn1g = FIN(48); const float* bn1b = FIN(49);
  const float* W2 = FIN(50); const float* b2 = FIN(51);
  const float* bn2g = FIN(52); const float* bn2b = FIN(53);
  const float* W3 = FIN(54); const float* b3 = FIN(55);
  const float* W4 = FIN(56); const float* b4 = FIN(57);
#undef FIN

  // workspace bump allocator (256B aligned)
  char* wsp = (char*)d_ws;
  size_t off = 0;
  auto alloc = [&](size_t bytes) -> void* {
    void* p = wsp + off;
    off = (off + bytes + 255) & ~(size_t)255;
    return p;
  };
  const size_t ND = (size_t)BATCH * MAXD;   // 1600
  const size_t NP = (size_t)BATCH * MAXP;   // 17440

  _Float16* wqT  = (_Float16*)alloc((size_t)EMB * EMB * 2);
  _Float16* wkT  = (_Float16*)alloc((size_t)EMB * EMB * 2);
  _Float16* wvT  = (_Float16*)alloc((size_t)EMB * EMB * 2);
  _Float16* woT  = (_Float16*)alloc((size_t)EMB * EMB * 2);
  _Float16* wiT  = (_Float16*)alloc((size_t)INTER * EMB * 2);
  _Float16* wo2T = (_Float16*)alloc((size_t)EMB * INTER * 2);
  _Float16* w1T  = (_Float16*)alloc((size_t)512 * FLATN * 2);
  _Float16* w2T  = (_Float16*)alloc((size_t)64 * 512 * 2);
  _Float16* w3T  = (_Float16*)alloc((size_t)32 * 64 * 2);
  _Float16* w4T  = (_Float16*)alloc((size_t)1 * 32 * 2);

  float*    xd32 = (float*)alloc(ND * EMB * 4);
  _Float16* xd16 = (_Float16*)alloc(ND * EMB * 2);
  float*    xp32 = (float*)alloc(NP * EMB * 4);
  _Float16* xp16 = (_Float16*)alloc(NP * EMB * 2);
  _Float16* q16  = (_Float16*)alloc(NP * EMB * 2);
  _Float16* k16  = (_Float16*)alloc(NP * EMB * 2);
  _Float16* v16  = (_Float16*)alloc(NP * EMB * 2);
  _Float16* ctx16= (_Float16*)alloc(NP * EMB * 2);
  float*    y32  = (float*)alloc(NP * EMB * 4);
  _Float16* h16  = (_Float16*)alloc(NP * INTER * 2);
  float*    I32  = (float*)alloc((size_t)BATCH * MAXD * MAXP * 4);
  _Float16* cf16 = (_Float16*)alloc((size_t)BATCH * FLATN * 2);
  float*    fc1_32 = (float*)alloc((size_t)BATCH * 512 * 4);
  _Float16* fc1_16 = (_Float16*)alloc((size_t)BATCH * 512 * 2);
  float*    fc2_32 = (float*)alloc((size_t)BATCH * 64 * 4);
  _Float16* fc2_16 = (_Float16*)alloc((size_t)BATCH * 64 * 2);
  _Float16* fc3_16 = (_Float16*)alloc((size_t)BATCH * 32 * 2);

  // 1. Embeddings + LN
  embed_ln_kernel<<<(int)ND, 128, 0, stream>>>(d_ids, d_word, d_pos, d_lng, d_lnb, xd32, xd16, MAXD);
  embed_ln_kernel<<<(int)NP, 128, 0, stream>>>(p_ids, p_word, p_pos, p_lng, p_lnb, xp32, xp16, MAXP);

  // 2. Transformer encoders (2 layers each), all GEMMs via WMMA f16
  run_encoder(stream, denc, xd32, xd16, MAXD, d_mask,
              wqT, wkT, wvT, woT, wiT, wo2T, q16, k16, v16, ctx16, y32, h16);
  run_encoder(stream, penc, xp32, xp16, MAXP, p_mask,
              wqT, wkT, wvT, woT, wiT, wo2T, q16, k16, v16, ctx16, y32, h16);

  // 3. Interaction map: I[b] = d_enc[b] (50x384) @ p_enc[b]^T -> (50,545)
  //    p_enc row-major (pos, emb) is exactly the Bt (NxK) layout needed.
  launch_gemm(stream, xd16, xp16, nullptr, I32, nullptr,
              MAXD, MAXP, EMB, BATCH,
              (long)MAXD * EMB, (long)MAXP * EMB, (long)MAXD * MAXP, 0);

  // 4. conv2d(1->3, k3, valid) + flatten -> f16
  {
    int tot = BATCH * 3 * 48 * 543;
    conv3_kernel<<<(tot + 255) / 256, 256, 0, stream>>>(I32, conv_w, conv_b, cf16);
  }

  // 5. FC stack. FC1 (K=78192) uses split-K (64 slices -> 1024 waves) with
  //    atomic f32 accumulation; ReLU is applied inside the following BN.
  launch_transp(stream, W1, w1T, FLATN, 512);
  hipMemsetAsync(fc1_32, 0, (size_t)BATCH * 512 * 4, stream);
  launch_gemm(stream, cf16, w1T, b1, fc1_32, nullptr, BATCH, 512, FLATN, 1,
              0, 0, 0, 0, /*ksplit=*/64);
  bn_kernel<<<(512 + 127) / 128, 128, 0, stream>>>(fc1_32, bn1g, bn1b, fc1_16, 512, 1);

  launch_transp(stream, W2, w2T, 512, 64);
  launch_gemm(stream, fc1_16, w2T, b2, fc2_32, nullptr, BATCH, 64, 512, 1, 0, 0, 0, 1);
  bn_kernel<<<1, 128, 0, stream>>>(fc2_32, bn2g, bn2b, fc2_16, 64, 0);

  launch_transp(stream, W3, w3T, 64, 32);
  launch_gemm(stream, fc2_16, w3T, b3, nullptr, fc3_16, BATCH, 32, 64, 1, 0, 0, 0, 1);

  launch_transp(stream, W4, w4T, 32, 1);
  launch_gemm(stream, fc3_16, w4T, b4, (float*)d_out, nullptr, BATCH, 1, 32, 1, 0, 0, 0, 0);
}
